// GATBlock_15530601742626
// MI455X (gfx1250) — matrix operand order repaired and checked by reading, not compiled, not run on hardware
//
#include <hip/hip_runtime.h>

// ---------------------------------------------------------------------------
// GATv2 block for MI455X (gfx1250, wave32).
//  * GEMMs: v_wmma_f32_16x16x32_bf16 with pre-converted bf16 operands
//    (B matrices pre-transposed so every fragment VGPR is one b32 load).
//  * Edge phase: wave-per-edge, double-buffered global_load_async_to_lds_b128
//    pipeline (ASYNCcnt) for the 2KB row gathers; f32 atomic scatter-add;
//    sortable-uint atomic segment max.
// ---------------------------------------------------------------------------

typedef __attribute__((ext_vector_type(16))) __bf16 v16bf;
typedef __attribute__((ext_vector_type(8)))  float  v8f;

union AF { v16bf bf; unsigned u[8]; };

#define D_DIM   128
#define HC      512
#define NEG_SL  0.2f
#define LN_EPS  1e-5f
#define SM_EPS  1e-16f

static __device__ __forceinline__ unsigned short bfbits(float f) {
  unsigned u = __float_as_uint(f);
  u += 0x7FFFu + ((u >> 16) & 1u);          // round-to-nearest-even
  return (unsigned short)(u >> 16);
}

// Monotonic float <-> uint mapping so atomicMax(u32) == float max. 0u == -inf.
static __device__ __forceinline__ unsigned fenc(float f) {
  unsigned u = __float_as_uint(f);
  return (u & 0x80000000u) ? ~u : (u | 0x80000000u);
}
static __device__ __forceinline__ float fdec(unsigned u) {
  unsigned b = (u & 0x80000000u) ? (u & 0x7FFFFFFFu) : ~u;
  return __uint_as_float(b);
}

// ------------------------------- utility kernels ---------------------------

__global__ void k_zero(float* __restrict__ p, size_t n) {
  for (size_t i = blockIdx.x * (size_t)blockDim.x + threadIdx.x; i < n;
       i += (size_t)gridDim.x * blockDim.x)
    p[i] = 0.0f;
}

__global__ void k_easum(const float* __restrict__ ea, float* __restrict__ easum, int E) {
  float p = 0.0f;
  for (int i = blockIdx.x * blockDim.x + threadIdx.x; i < E;
       i += gridDim.x * blockDim.x)
    p += ea[i];
  #pragma unroll
  for (int off = 16; off; off >>= 1) p += __shfl_xor(p, off);
  if ((threadIdx.x & 31) == 0) atomicAdd(easum, p);
}

// f32 -> bf16, same layout (used for x: A-fragment K-pairs are contiguous)
__global__ void k_cvt(const float* __restrict__ src, unsigned short* __restrict__ dst,
                      int n) {
  for (int i = blockIdx.x * blockDim.x + threadIdx.x; i < n;
       i += gridDim.x * blockDim.x)
    dst[i] = bfbits(src[i]);
}

// f32 [K, Ncols] -> bf16 transposed [Ncols, K]: B-fragment K-pairs contiguous
__global__ void k_cvt_wT(const float* __restrict__ W, unsigned short* __restrict__ WT,
                         int K, int Ncols) {
  const int tot = K * Ncols;
  for (int i = blockIdx.x * blockDim.x + threadIdx.x; i < tot;
       i += gridDim.x * blockDim.x) {
    const int n = i / K;
    const int k = i - n * K;
    WT[i] = bfbits(W[(size_t)k * Ncols + n]);
  }
}

// bias_proj[col] = sum_k bias_out[k] * Wp[k,col]  (folds bias_out @ Wp)
__global__ void k_bias_proj(const float* __restrict__ bias_out,
                            const float* __restrict__ Wp,
                            float* __restrict__ bproj) {
  const int col = threadIdx.x;           // 128 threads
  float s = 0.0f;
  for (int k = 0; k < HC; ++k) s += bias_out[k] * Wp[(size_t)k * D_DIM + col];
  bproj[col] = s;
}

// ------------------------ GEMM1: xl = x@Wl+bl, xr = x@Wr+br ----------------
// One wave -> one 16x16 tile of BOTH xl and xr (shared A fragment).
// All operands pre-converted bf16; fragments are straight b32 loads.

__global__ void k_gemm_xlxr(const unsigned short* __restrict__ xbf,
                            const unsigned short* __restrict__ WlT,
                            const unsigned short* __restrict__ WrT,
                            const float* __restrict__ bl, const float* __restrict__ br,
                            float* __restrict__ xl, float* __restrict__ xr, int N) {
  const int lane = threadIdx.x & 31;
  const int wave = threadIdx.x >> 5;
  const int row0 = blockIdx.x * 16;
  const int col0 = (blockIdx.y * 4 + wave) * 16;
  const int half = lane >> 4;
  const int l16  = lane & 15;

  int arow = row0 + l16;
  if (arow > N - 1) arow = N - 1;
  const unsigned short* __restrict__ ap = xbf + (size_t)arow * D_DIM;
  const unsigned short* __restrict__ blp = WlT + (size_t)(col0 + l16) * D_DIM;
  const unsigned short* __restrict__ brp = WrT + (size_t)(col0 + l16) * D_DIM;

  v8f accl = {}; v8f accr = {};
  #pragma unroll
  for (int kt = 0; kt < 4; ++kt) {
    const int kb = kt * 32;
    AF a, bL, bR;
    #pragma unroll
    for (int v = 0; v < 8; ++v) {
      const int k0 = kb + (v >> 2) * 16 + half * 8 + (v & 3) * 2;  // A pair
      const int kk = kb + (v >> 2) * 16 + half * 8 + (v & 3) * 2;  // B pair
      a.u[v]  = *(const unsigned*)(ap  + k0);
      bL.u[v] = *(const unsigned*)(blp + kk);
      bR.u[v] = *(const unsigned*)(brp + kk);
    }
    accl = __builtin_amdgcn_wmma_f32_16x16x32_bf16(false, a.bf, false, bL.bf, (short)0, accl, false, false);
    accr = __builtin_amdgcn_wmma_f32_16x16x32_bf16(false, a.bf, false, bR.bf, (short)0, accr, false, false);
  }
  #pragma unroll
  for (int r = 0; r < 8; ++r) {
    const int row = row0 + r + half * 8;
    const int col = col0 + l16;
    if (row < N) {
      xl[(size_t)row * HC + col] = accl[r] + bl[col];
      xr[(size_t)row * HC + col] = accr[r] + br[col];
    }
  }
}

// ---------------- edge pass A: logits + atomic segment max -----------------
// One wave per edge (grid-stride). Double-buffered async gather of the
// xl[src]/xr[dst] rows into LDS via global_load_async_to_lds_b128 while the
// previous edge is being reduced; s_wait_asynccnt gates buffer handoff.
// NOTE: smem is the only LDS object in this kernel -> LDS offset 0.

__global__ void __launch_bounds__(128)
k_edge_logits(const int* __restrict__ eidx,
              const float* __restrict__ eattr,
              const float* __restrict__ We,
              const float* __restrict__ att,
              const float* __restrict__ xl,
              const float* __restrict__ xr,
              const float* __restrict__ easum,
              float* __restrict__ logits,
              unsigned* __restrict__ mbuf,
              int E, int EP) {
  __shared__ float smem[4][2][1024];   // [wave][buf][xl row | xr row], 32 KB
  const int lane = threadIdx.x & 31;
  const int wv   = threadIdx.x >> 5;
  const int w  = (blockIdx.x * blockDim.x + threadIdx.x) >> 5;
  const int nw = (gridDim.x * blockDim.x) >> 5;
  const float eam = easum[0] / (float)E;
  const int* __restrict__ srcp = eidx;
  const int* __restrict__ dstp = eidx + E;

  float wei[16], ati[16];
  #pragma unroll
  for (int t = 0; t < 16; ++t) {
    wei[t] = We[lane + 32 * t];
    ati[t] = att[lane + 32 * t];
  }

  if (w >= EP) return;

  const unsigned wbase = (unsigned)wv * 8192u + (unsigned)lane * 16u;

  // issue the 8 async b128 loads (xl row + xr row) for edge e into buffer buf
  auto issue = [&](int e, int buf) {
    const int s = (e < E) ? srcp[e] : (e - E);
    const int d = (e < E) ? dstp[e] : (e - E);
    const unsigned long long gl =
        (unsigned long long)(uintptr_t)(xl + (size_t)s * HC) + (unsigned long long)lane * 16u;
    const unsigned long long gr =
        (unsigned long long)(uintptr_t)(xr + (size_t)d * HC) + (unsigned long long)lane * 16u;
    const unsigned lb = wbase + (unsigned)buf * 4096u;
    #pragma unroll
    for (int c = 0; c < 4; ++c) {
      const unsigned          la = lb + (unsigned)c * 512u;
      const unsigned long long ga = gl + (unsigned long long)c * 512u;
      asm volatile("global_load_async_to_lds_b128 %0, %1, off"
                   :: "v"(la), "v"(ga) : "memory");
    }
    #pragma unroll
    for (int c = 0; c < 4; ++c) {
      const unsigned          la = lb + 2048u + (unsigned)c * 512u;
      const unsigned long long ga = gr + (unsigned long long)c * 512u;
      asm volatile("global_load_async_to_lds_b128 %0, %1, off"
                   :: "v"(la), "v"(ga) : "memory");
    }
  };

  issue(w, 0);
  int cur = 0;
  for (int e = w; e < EP; e += nw) {
    const int en = e + nw;
    if (en < EP) {
      issue(en, cur ^ 1);
      asm volatile("s_wait_asynccnt 0x8" ::: "memory");  // current buffer ready
    } else {
      asm volatile("s_wait_asynccnt 0x0" ::: "memory");
    }

    const int   d   = (e < E) ? dstp[e]  : (e - E);
    const float eav = (e < E) ? eattr[e] : eam;
    const float* __restrict__ pl = &smem[wv][cur][0];
    const float* __restrict__ pr = &smem[wv][cur][512];

    float a0 = 0.f, a1 = 0.f, a2 = 0.f, a3 = 0.f;
    #pragma unroll
    for (int t = 0; t < 16; ++t) {
      const int i = lane + 32 * t;               // head = t>>2
      float z = pl[i] + pr[i] + eav * wei[t];
      z = (z > 0.f) ? z : NEG_SL * z;
      const float p = z * ati[t];
      if      (t < 4)  a0 += p;
      else if (t < 8)  a1 += p;
      else if (t < 12) a2 += p;
      else             a3 += p;
    }
    #pragma unroll
    for (int off = 16; off; off >>= 1) {
      a0 += __shfl_xor(a0, off);
      a1 += __shfl_xor(a1, off);
      a2 += __shfl_xor(a2, off);
      a3 += __shfl_xor(a3, off);
    }
    if (lane == 0) {
      float ah[4] = { a0, a1, a2, a3 };
      #pragma unroll
      for (int h = 0; h < 4; ++h) {
        logits[(size_t)e * 4 + h] = ah[h];
        atomicMax(mbuf + (size_t)d * 4 + h, fenc(ah[h]));
      }
    }
    cur ^= 1;
  }
}

// ---------------- edge pass B: exp(alpha - max), atomic segment sum --------

__global__ void k_exp_sum(const int* __restrict__ eidx,
                          const unsigned* __restrict__ mbuf,
                          float* __restrict__ logits,
                          float* __restrict__ ssum,
                          int E, int EP) {
  const int tot = EP * 4;
  for (int tid = blockIdx.x * blockDim.x + threadIdx.x; tid < tot;
       tid += gridDim.x * blockDim.x) {
    const int e = tid >> 2;
    const int h = tid & 3;
    const int d = (e < E) ? eidx[E + e] : (e - E);
    const float mm = fdec(mbuf[(size_t)d * 4 + h]);
    const float ev = expf(logits[tid] - mm);
    logits[tid] = ev;
    atomicAdd(ssum + (size_t)d * 4 + h, ev);
  }
}

// ------------- edge pass C: normalized weighted scatter-add ----------------
// One wave per edge; 512 coalesced global_atomic_add_f32 into out[dst].

__global__ void k_scatter(const int* __restrict__ eidx,
                          const float* __restrict__ expv,
                          const float* __restrict__ ssum,
                          const float* __restrict__ xl,
                          float* __restrict__ outac,
                          int E, int EP) {
  const int lane = threadIdx.x & 31;
  const int w  = (blockIdx.x * blockDim.x + threadIdx.x) >> 5;
  const int nw = (gridDim.x * blockDim.x) >> 5;
  const int* __restrict__ srcp = eidx;
  const int* __restrict__ dstp = eidx + E;

  for (int e = w; e < EP; e += nw) {
    const int s = (e < E) ? srcp[e] : (e - E);
    const int d = (e < E) ? dstp[e] : (e - E);

    const int en = e + nw;
    if (en < EP) {
      const int sn = (en < E) ? srcp[en] : (en - E);
      __builtin_prefetch(xl + (size_t)sn * HC + lane * 16, 0, 0);
    }

    float wh[4];
    #pragma unroll
    for (int h = 0; h < 4; ++h)
      wh[h] = expv[(size_t)e * 4 + h] / (ssum[(size_t)d * 4 + h] + SM_EPS);

    const float* __restrict__ pl = xl + (size_t)s * HC;
    float* __restrict__ po = outac + (size_t)d * HC;
    #pragma unroll
    for (int t = 0; t < 16; ++t) {
      const int i = lane + 32 * t;
      atomicAdd(po + i, wh[t >> 2] * pl[i]);
    }
  }
}

// ------------- GEMM2: y = gelu(out@Wp + bias_proj + bp + x) ----------------
// K = 512 -> 16 WMMA steps. A is the f32 scatter accumulator (converted
// inline); B is pre-transposed bf16. bias_out@Wp folded into bias_proj.

__global__ void k_gemm_proj(const float* __restrict__ acc_in,
                            const unsigned short* __restrict__ WpT,
                            const float* __restrict__ bproj,
                            const float* __restrict__ bp,
                            const float* __restrict__ x,
                            float* __restrict__ ytmp, int N) {
  const int lane = threadIdx.x & 31;
  const int wave = threadIdx.x >> 5;
  const int row0 = blockIdx.x * 16;
  const int col0 = (blockIdx.y * 4 + wave) * 16;
  const int half = lane >> 4;
  const int l16  = lane & 15;

  int arow = row0 + l16;
  if (arow > N - 1) arow = N - 1;
  const float* __restrict__ ap = acc_in + (size_t)arow * HC;
  const unsigned short* __restrict__ bpp = WpT + (size_t)(col0 + l16) * HC;

  v8f acc = {};
  #pragma unroll 4
  for (int kt = 0; kt < 16; ++kt) {
    const int kb = kt * 32;
    AF a, b;
    #pragma unroll
    for (int v = 0; v < 8; ++v) {
      const int k0 = kb + (v >> 2) * 16 + half * 8 + (v & 3) * 2;
      const int kk = kb + (v >> 2) * 16 + half * 8 + (v & 3) * 2;
      const unsigned lo = bfbits(ap[k0]);
      const unsigned hi = bfbits(ap[k0 + 1]);
      a.u[v] = lo | (hi << 16);
      b.u[v] = *(const unsigned*)(bpp + kk);
    }
    acc = __builtin_amdgcn_wmma_f32_16x16x32_bf16(false, a.bf, false, b.bf, (short)0, acc, false, false);
  }
  #pragma unroll
  for (int r = 0; r < 8; ++r) {
    const int row = row0 + r + half * 8;
    const int col = col0 + l16;
    if (row < N) {
      float v = acc[r] + bproj[col] + bp[col] + x[(size_t)row * D_DIM + col];
      float g = 0.5f * v * (1.0f + erff(v * 0.70710678118654752f));  // exact gelu
      ytmp[(size_t)row * D_DIM + col] = g;
    }
  }
}

// ---------------------------- LayerNorm (wave/row) -------------------------

__global__ void k_layernorm(const float* __restrict__ y,
                            const float* __restrict__ gamma,
                            const float* __restrict__ beta,
                            float* __restrict__ out, int N) {
  const int lane = threadIdx.x & 31;
  const int w  = (blockIdx.x * blockDim.x + threadIdx.x) >> 5;
  const int nw = (gridDim.x * blockDim.x) >> 5;
  for (int row = w; row < N; row += nw) {
    const float* __restrict__ pr = y + (size_t)row * D_DIM;
    const float v0 = pr[lane], v1 = pr[lane + 32], v2 = pr[lane + 64], v3 = pr[lane + 96];
    float s  = v0 + v1 + v2 + v3;
    float sq = v0 * v0 + v1 * v1 + v2 * v2 + v3 * v3;
    #pragma unroll
    for (int off = 16; off; off >>= 1) {
      s  += __shfl_xor(s, off);
      sq += __shfl_xor(sq, off);
    }
    const float mu  = s * (1.0f / 128.0f);
    const float var = sq * (1.0f / 128.0f) - mu * mu;
    const float rs  = rsqrtf(var + LN_EPS);
    float* __restrict__ po = out + (size_t)row * D_DIM;
    po[lane]      = (v0 - mu) * rs * gamma[lane]      + beta[lane];
    po[lane + 32] = (v1 - mu) * rs * gamma[lane + 32] + beta[lane + 32];
    po[lane + 64] = (v2 - mu) * rs * gamma[lane + 64] + beta[lane + 64];
    po[lane + 96] = (v3 - mu) * rs * gamma[lane + 96] + beta[lane + 96];
  }
}

// ------------------------------- host launcher -----------------------------

extern "C" void kernel_launch(void* const* d_in, const int* in_sizes, int n_in,
                              void* d_out, int out_size, void* d_ws, size_t ws_size,
                              hipStream_t stream) {
  const float* x        = (const float*)d_in[0];
  const int*   eidx     = (const int*)d_in[1];      // integer inputs arrive as int32
  const float* eattr    = (const float*)d_in[2];
  const float* Wl       = (const float*)d_in[3];
  const float* bl       = (const float*)d_in[4];
  const float* Wr       = (const float*)d_in[5];
  const float* br       = (const float*)d_in[6];
  const float* We       = (const float*)d_in[7];
  const float* att      = (const float*)d_in[8];
  const float* bias_out = (const float*)d_in[9];
  const float* Wp       = (const float*)d_in[10];
  const float* bp       = (const float*)d_in[11];
  const float* gamma    = (const float*)d_in[12];
  const float* beta     = (const float*)d_in[13];
  float* out = (float*)d_out;

  const int N  = in_sizes[0] / D_DIM;   // 10000
  const int E  = in_sizes[1] / 2;       // 160000
  const int EP = E + N;                 // with self loops

  // ---- workspace layout: zeroed accumulators first, then f32, then bf16 ----
  float* ws = (float*)d_ws;
  size_t off = 0;
  float*    outac  = ws + off;               off += (size_t)N * HC;   // scatter accum
  unsigned* mbuf   = (unsigned*)(ws + off);  off += (size_t)N * 4;    // segment max (enc)
  float*    ssum   = ws + off;               off += (size_t)N * 4;    // segment sum
  float*    easum  = ws + off;               off += 1;                // edge_attr sum
  const size_t zcount = off;
  off = (off + 255) & ~(size_t)255;
  float*    xl     = ws + off;               off += (size_t)N * HC;
  float*    xr     = ws + off;               off += (size_t)N * HC;
  float*    logits = ws + off;               off += (size_t)EP * 4;   // reused as expv
  float*    bproj  = ws + off;               off += 256;
  float*    ytmp   = xr;                     // xr dead after logits pass

  unsigned short* xbf = (unsigned short*)(ws + off);
  unsigned short* WlT = xbf + (size_t)N * D_DIM;       // [HC][D] bf16, transposed
  unsigned short* WrT = WlT + (size_t)HC * D_DIM;
  unsigned short* WpT = WrT + (size_t)HC * D_DIM;      // [D][HC] bf16, transposed

  // 1) zero accumulators (graph-replay safe re-init)
  {
    int blocks = (int)((zcount + 255) / 256);
    if (blocks > 4096) blocks = 4096;
    k_zero<<<blocks, 256, 0, stream>>>(ws, zcount);
  }
  // 2) mean of edge_attr (scalar, ED==1)
  k_easum<<<128, 256, 0, stream>>>(eattr, easum, E);
  // 3) bf16 pre-conversion of static operands
  k_cvt<<<(N * D_DIM + 255) / 256, 256, 0, stream>>>(x, xbf, N * D_DIM);
  k_cvt_wT<<<256, 256, 0, stream>>>(Wl, WlT, D_DIM, HC);
  k_cvt_wT<<<256, 256, 0, stream>>>(Wr, WrT, D_DIM, HC);
  k_cvt_wT<<<256, 256, 0, stream>>>(Wp, WpT, HC, D_DIM);
  k_bias_proj<<<1, 128, 0, stream>>>(bias_out, Wp, bproj);
  // 4) xl / xr GEMMs (WMMA bf16)
  {
    dim3 g((N + 15) / 16, HC / 64);   // (625, 8), 4 waves/block
    k_gemm_xlxr<<<g, 128, 0, stream>>>(xbf, WlT, WrT, bl, br, xl, xr, N);
  }
  // 5) edge logits + segment max (async-LDS pipelined gather)
  k_edge_logits<<<2048, 128, 0, stream>>>(eidx, eattr, We, att, xl, xr, easum,
                                          logits, mbuf, E, EP);
  // 6) exp + segment sum
  k_exp_sum<<<(EP * 4 + 255) / 256, 256, 0, stream>>>(eidx, mbuf, logits, ssum, E, EP);
  // 7) normalized weighted scatter-add
  k_scatter<<<1024, 256, 0, stream>>>(eidx, logits, ssum, xl, outac, E, EP);
  // 8) projection GEMM + residual + gelu (WMMA bf16)
  {
    dim3 g((N + 15) / 16, D_DIM / 64); // (625, 2)
    k_gemm_proj<<<g, 128, 0, stream>>>(outac, WpT, bproj, bp, x, ytmp, N);
  }
  // 9) layernorm -> d_out
  k_layernorm<<<1280, 256, 0, stream>>>(ytmp, gamma, beta, out, N);
}
